// MultiHeadLatentAttention_17308718203689
// MI455X (gfx1250) — matrix operand order repaired and checked by reading, not compiled, not run on hardware
//
#include <hip/hip_runtime.h>

// ---------------------------------------------------------------------------
// MLA forward for MI455X (gfx1250, wave32, WMMA + async global->LDS copies).
// Pipeline: fp32->bf16 convert, 3 bf16 WMMA GEMMs (q / latent / kv-up),
// flash-attention with WMMA QK^T and PV, final WMMA GEMM -> fp32 out.
// ~163 GFLOP vs ~100MB traffic => compute bound => matrix cores, and
// double-buffered LDS tiles fed by GLOBAL_LOAD_ASYNC_TO_LDS_B128 (ASYNCcnt).
// ---------------------------------------------------------------------------

typedef __bf16 bf16;
typedef __attribute__((ext_vector_type(16))) __bf16 v16bf;
typedef __attribute__((ext_vector_type(8)))  __bf16 v8bf;
typedef __attribute__((ext_vector_type(4)))  __bf16 v4bf;
typedef __attribute__((ext_vector_type(8)))  float  v8f;

#define EMBED  2048
#define NHEADS 16
#define HDIM   128
#define LATENT 512
#define SEQ    2048
#define BATCH  2
#define ROWS   (BATCH * SEQ)   // 4096 token rows

// Per-lane 16B async copy global -> LDS (no VGPR round trip, ASYNCcnt).
// lds_addr: raw LDS byte address (low 32 bits of generic pointer);
// gbase:    uniform 64-bit global base (SGPR pair);
// goff:     per-lane byte offset (VGPR).
__device__ __forceinline__ void async_copy16(unsigned lds_addr,
                                             unsigned long long gbase,
                                             unsigned goff) {
    asm volatile("global_load_async_to_lds_b128 %0, %1, %2"
                 :
                 : "v"(lds_addr), "v"(goff), "s"(gbase)
                 : "memory");
}
__device__ __forceinline__ void wait_async0() {
    asm volatile("s_wait_asynccnt 0" ::: "memory");
}

// Build a 16-element bf16 WMMA operand from two contiguous 16B LDS chunks.
// A-fragment (16x32, 16-bit): lane half h, elems j<8 -> k=8h+j, j>=8 -> k=16+8h+(j-8)
//   => chunks at [k0+8h] and [k0+16+8h].
// B-fragment (32x16, 16-bit): same k order as A
//   => chunks at [k0+8h] and [k0+16+8h].  (tile stored (n,k)-major)
__device__ __forceinline__ v16bf load_frag(const bf16* p0, const bf16* p1) {
    v8bf lo = *(const v8bf*)p0;
    v8bf hi = *(const v8bf*)p1;
    return __builtin_shufflevector(lo, hi, 0,1,2,3,4,5,6,7,8,9,10,11,12,13,14,15);
}

// ---------------------------------------------------------------------------
// fp32 -> bf16 conversion (vectorized x4)
// ---------------------------------------------------------------------------
__global__ __launch_bounds__(256)
void cvt_bf16(const float* __restrict__ in, bf16* __restrict__ out, int n4) {
    int i = blockIdx.x * 256 + threadIdx.x;
    if (i < n4) {
        float4 f = ((const float4*)in)[i];
        v4bf o;
        o[0] = (bf16)f.x; o[1] = (bf16)f.y; o[2] = (bf16)f.z; o[3] = (bf16)f.w;
        ((v4bf*)out)[i] = o;
    }
}

// ---------------------------------------------------------------------------
// GEMM: C[M,N] = A[M,K] @ B[N,K]^T + bias[N]
// Block: 256 threads = 8 waves; tile 128x128, BK=64, double-buffered LDS.
// Wave grid 2x4: each wave -> 64 rows x 32 cols = 4x2 WMMA accumulators.
// Tile k+1 is prefetched with async copies while tile k is computed.
// ---------------------------------------------------------------------------
template <bool OUT_F32>
__global__ __launch_bounds__(256)
void gemm_bias(const bf16* __restrict__ A, const bf16* __restrict__ B,
               const float* __restrict__ bias, void* __restrict__ Cout,
               int M, int N, int K) {
    __shared__ bf16 sA[2][128][72];   // (m, k) rows contiguous, +8 pad
    __shared__ bf16 sB[2][128][72];   // (n, k) rows contiguous, +8 pad

    const int tid  = threadIdx.x;
    const int wave = tid >> 5;
    const int lane = tid & 31;
    const int h    = lane >> 4;    // lane half
    const int l16  = lane & 15;
    const int wm   = wave >> 2;    // 0..1
    const int wn   = wave & 3;     // 0..3
    const int m0   = blockIdx.y * 128;
    const int n0   = blockIdx.x * 128;

    // Per-thread staging coords: 128 rows x 8 chunks (8 bf16) = 1024 chunks
    // per matrix; 4 chunks per thread per matrix.
    unsigned aoff[4], boff[4];
    unsigned ldsA[2][4], ldsB[2][4];
#pragma unroll
    for (int i = 0; i < 4; i++) {
        int cid = tid + i * 256;
        int r   = cid >> 3;
        int cc  = (cid & 7) * 8;
        aoff[i] = (unsigned)(((m0 + r) * K + cc) * 2);
        boff[i] = (unsigned)(((n0 + r) * K + cc) * 2);
        ldsA[0][i] = (unsigned)(size_t)&sA[0][r][cc];
        ldsA[1][i] = (unsigned)(size_t)&sA[1][r][cc];
        ldsB[0][i] = (unsigned)(size_t)&sB[0][r][cc];
        ldsB[1][i] = (unsigned)(size_t)&sB[1][r][cc];
    }
    const unsigned long long Ab = (unsigned long long)(size_t)A;
    const unsigned long long Bb = (unsigned long long)(size_t)B;

    v8f acc[4][2];
#pragma unroll
    for (int i = 0; i < 4; i++)
#pragma unroll
        for (int j = 0; j < 2; j++) acc[i][j] = (v8f){0,0,0,0,0,0,0,0};

    // Prologue: async-copy tile 0 into buffer 0.
#pragma unroll
    for (int i = 0; i < 4; i++) {
        async_copy16(ldsA[0][i], Ab, aoff[i]);
        async_copy16(ldsB[0][i], Bb, boff[i]);
    }
    wait_async0();
    __syncthreads();

    int cur = 0;
    for (int k0 = 0; k0 < K; k0 += 64) {
        const int nxt = cur ^ 1;
        if (k0 + 64 < K) {
            unsigned kb = (unsigned)((k0 + 64) * 2);
#pragma unroll
            for (int i = 0; i < 4; i++) {
                async_copy16(ldsA[nxt][i], Ab, aoff[i] + kb);
                async_copy16(ldsB[nxt][i], Bb, boff[i] + kb);
            }
        }

#pragma unroll
        for (int kk = 0; kk < 64; kk += 32) {
            v16bf afr[4], bfr[2];
#pragma unroll
            for (int mt = 0; mt < 4; mt++) {
                int r = wm * 64 + mt * 16 + l16;
                afr[mt] = load_frag(&sA[cur][r][kk + 8 * h],
                                    &sA[cur][r][kk + 16 + 8 * h]);
            }
#pragma unroll
            for (int nt = 0; nt < 2; nt++) {
                int c = wn * 32 + nt * 16 + l16;
                bfr[nt] = load_frag(&sB[cur][c][kk + 8 * h],
                                    &sB[cur][c][kk + 16 + 8 * h]);
            }
#pragma unroll
            for (int mt = 0; mt < 4; mt++)
#pragma unroll
                for (int nt = 0; nt < 2; nt++)
                    acc[mt][nt] = __builtin_amdgcn_wmma_f32_16x16x32_bf16(
                        false, afr[mt], false, bfr[nt], (short)0, acc[mt][nt], false, false);
        }

        wait_async0();
        __syncthreads();
        cur = nxt;
    }

    // Epilogue: C layout -> lane holds m = 8h+g, n = l16 (+16 per n-tile)
#pragma unroll
    for (int mt = 0; mt < 4; mt++) {
        int mbase = m0 + wm * 64 + mt * 16 + 8 * h;
#pragma unroll
        for (int nt = 0; nt < 2; nt++) {
            int n = n0 + wn * 32 + nt * 16 + l16;
            float bv = bias[n];
#pragma unroll
            for (int g = 0; g < 8; g++) {
                float val = acc[mt][nt][g] + bv;
                if (OUT_F32)
                    ((float*)Cout)[(size_t)(mbase + g) * N + n] = val;
                else
                    ((bf16*)Cout)[(size_t)(mbase + g) * N + n] = (bf16)val;
            }
        }
    }
}

// ---------------------------------------------------------------------------
// Flash attention: one block handles 128 queries of one (batch, head).
// 8 waves, each owns 16 full query rows (softmax stats stay intra-halfwave).
// Loops over 64-key blocks: S = Q K^T (WMMA), online softmax, O += P V (WMMA).
// Q and K tiles staged with async global->LDS copies; V goes through VGPRs
// because it is stored transposed (d, key) for the PV B-fragments.
// ---------------------------------------------------------------------------
__global__ __launch_bounds__(256)
void mla_attn(const bf16* __restrict__ Q, const bf16* __restrict__ KV,
              bf16* __restrict__ O) {
    __shared__ bf16 sQ[128][136];     // (q, d)
    __shared__ bf16 sK[64][136];      // (key, d)
    __shared__ bf16 sVt[128][72];     // (d, key)  transposed for PV B-frags
    __shared__ bf16 sP[8][16][72];    // per-wave P tile (m_local, key)

    const int tid  = threadIdx.x;
    const int wave = tid >> 5;
    const int lane = tid & 31;
    const int h    = lane >> 4;
    const int l16  = lane & 15;
    const int q0   = blockIdx.x * 128;
    const int head = blockIdx.y;
    const int bat  = blockIdx.z;

    const size_t qbase  = (size_t)bat * SEQ * EMBED + (size_t)head * HDIM;
    const size_t kvbase = (size_t)bat * SEQ * (2 * EMBED) + (size_t)head * HDIM;
    const unsigned long long Qb  = (unsigned long long)(size_t)Q;
    const unsigned long long KVb = (unsigned long long)(size_t)KV;

    // Async-load Q tile: 128x128 bf16 = 2048 chunks of 8; 8 per thread
#pragma unroll
    for (int i = 0; i < 8; i++) {
        int cid = tid + i * 256;
        int r   = cid >> 4;
        int cc  = (cid & 15) * 8;
        async_copy16((unsigned)(size_t)&sQ[r][cc], Qb,
                     (unsigned)((qbase + (size_t)(q0 + r) * EMBED + cc) * 2));
    }

    float rmax[8], rsum[8];
#pragma unroll
    for (int g = 0; g < 8; g++) { rmax[g] = -1e30f; rsum[g] = 0.f; }
    v8f accO[8];
#pragma unroll
    for (int i = 0; i < 8; i++) accO[i] = (v8f){0,0,0,0,0,0,0,0};

    const float scale = 0.08838834764831845f;   // 1/sqrt(128)

    for (int kb = 0; kb < SEQ; kb += 64) {
        __syncthreads();
        // K tile async (row-major); V tile transposed via VGPRs:
        // 64 keys x 16 chunks = 1024 chunks; 4 per thread
#pragma unroll
        for (int i = 0; i < 4; i++) {
            int cid = tid + i * 256;
            int r   = cid >> 4;            // key 0..63
            int cc  = (cid & 15) * 8;      // d chunk
            async_copy16((unsigned)(size_t)&sK[r][cc], KVb,
                         (unsigned)((kvbase + (size_t)(kb + r) * (2 * EMBED) + cc) * 2));
            v8bf vv = *(const v8bf*)&KV[kvbase + EMBED + (size_t)(kb + r) * (2 * EMBED) + cc];
#pragma unroll
            for (int e = 0; e < 8; e++) sVt[cc + e][r] = vv[e];
        }
        wait_async0();
        __syncthreads();

        // S = Q K^T : 16 query rows x 64 keys per wave
        v8f sacc[4];
#pragma unroll
        for (int nt = 0; nt < 4; nt++) sacc[nt] = (v8f){0,0,0,0,0,0,0,0};
#pragma unroll
        for (int kk = 0; kk < HDIM; kk += 32) {
            int r = wave * 16 + l16;
            v16bf aq = load_frag(&sQ[r][kk + 8 * h], &sQ[r][kk + 16 + 8 * h]);
#pragma unroll
            for (int nt = 0; nt < 4; nt++) {
                int c = nt * 16 + l16;
                v16bf bk = load_frag(&sK[c][kk + 8 * h], &sK[c][kk + 16 + 8 * h]);
                sacc[nt] = __builtin_amdgcn_wmma_f32_16x16x32_bf16(
                    false, aq, false, bk, (short)0, sacc[nt], false, false);
            }
        }
#pragma unroll
        for (int nt = 0; nt < 4; nt++)
#pragma unroll
            for (int g = 0; g < 8; g++) sacc[nt][g] *= scale;

        // Online softmax per row m = 8h + g (replicated across 16-lane half)
        float alpha[8];
#pragma unroll
        for (int g = 0; g < 8; g++) {
            float mv = fmaxf(fmaxf(sacc[0][g], sacc[1][g]),
                             fmaxf(sacc[2][g], sacc[3][g]));
#pragma unroll
            for (int m = 1; m < 16; m <<= 1) mv = fmaxf(mv, __shfl_xor(mv, m, 32));
            float nm = fmaxf(rmax[g], mv);
            alpha[g] = __expf(rmax[g] - nm);
            rmax[g]  = nm;
            float ps = 0.f;
#pragma unroll
            for (int nt = 0; nt < 4; nt++) {
                float p = __expf(sacc[nt][g] - nm);
                sP[wave][8 * h + g][nt * 16 + l16] = (bf16)p;
                ps += p;
            }
#pragma unroll
            for (int m = 1; m < 16; m <<= 1) ps += __shfl_xor(ps, m, 32);
            rsum[g] = rsum[g] * alpha[g] + ps;
#pragma unroll
            for (int nt2 = 0; nt2 < 8; nt2++) accO[nt2][g] *= alpha[g];
        }

        // O += P V   (wave-local sP; no cross-wave sync needed)
#pragma unroll
        for (int kk = 0; kk < 64; kk += 32) {
            v16bf ap = load_frag(&sP[wave][l16][kk + 8 * h],
                                 &sP[wave][l16][kk + 16 + 8 * h]);
#pragma unroll
            for (int nt2 = 0; nt2 < 8; nt2++) {
                int d = nt2 * 16 + l16;
                v16bf bv = load_frag(&sVt[d][kk + 8 * h], &sVt[d][kk + 16 + 8 * h]);
                accO[nt2] = __builtin_amdgcn_wmma_f32_16x16x32_bf16(
                    false, ap, false, bv, (short)0, accO[nt2], false, false);
            }
        }
    }

    // Normalize and write O as bf16 into (b*s, EMBED) with col = head*128 + d
#pragma unroll
    for (int nt2 = 0; nt2 < 8; nt2++) {
        int d = nt2 * 16 + l16;
#pragma unroll
        for (int g = 0; g < 8; g++) {
            int q = q0 + wave * 16 + 8 * h + g;
            O[(size_t)(bat * SEQ + q) * EMBED + head * HDIM + d] =
                (bf16)(accO[nt2][g] / rsum[g]);
        }
    }
}

// ---------------------------------------------------------------------------
extern "C" void kernel_launch(void* const* d_in, const int* in_sizes, int n_in,
                              void* d_out, int out_size, void* d_ws, size_t ws_size,
                              hipStream_t stream) {
    const float* x     = (const float*)d_in[0];
    const float* q_w   = (const float*)d_in[1];
    const float* q_b   = (const float*)d_in[2];
    const float* kvd_w = (const float*)d_in[3];
    const float* kvd_b = (const float*)d_in[4];
    const float* kvu_w = (const float*)d_in[5];
    const float* kvu_b = (const float*)d_in[6];
    const float* out_w = (const float*)d_in[7];
    const float* out_b = (const float*)d_in[8];

    char*  ws  = (char*)d_ws;
    size_t off = 0;
    auto alloc_bf16 = [&](size_t elems) {
        bf16* p = (bf16*)(ws + off);
        off += ((elems * sizeof(bf16) + 255) & ~(size_t)255);
        return p;
    };
    bf16* xb    = alloc_bf16((size_t)ROWS * EMBED);
    bf16* qwb   = alloc_bf16((size_t)EMBED * EMBED);
    bf16* kvdwb = alloc_bf16((size_t)LATENT * EMBED);
    bf16* kvuwb = alloc_bf16((size_t)2 * EMBED * LATENT);
    bf16* outwb = alloc_bf16((size_t)EMBED * EMBED);
    bf16* qb    = alloc_bf16((size_t)ROWS * EMBED);
    bf16* latb  = alloc_bf16((size_t)ROWS * LATENT);
    bf16* kvb   = alloc_bf16((size_t)ROWS * 2 * EMBED);
    bf16* attnb = alloc_bf16((size_t)ROWS * EMBED);

    auto cvt = [&](const float* in, bf16* out, size_t n) {
        int n4 = (int)(n / 4);
        cvt_bf16<<<(n4 + 255) / 256, 256, 0, stream>>>(in, out, n4);
    };
    cvt(x,     xb,    (size_t)ROWS * EMBED);
    cvt(q_w,   qwb,   (size_t)EMBED * EMBED);
    cvt(kvd_w, kvdwb, (size_t)LATENT * EMBED);
    cvt(kvu_w, kvuwb, (size_t)2 * EMBED * LATENT);
    cvt(out_w, outwb, (size_t)EMBED * EMBED);

    dim3 blk(256);
    // q = x @ q_w^T + q_b
    gemm_bias<false><<<dim3(EMBED / 128, ROWS / 128), blk, 0, stream>>>(
        xb, qwb, q_b, qb, ROWS, EMBED, EMBED);
    // latent = x @ kvd_w^T + kvd_b
    gemm_bias<false><<<dim3(LATENT / 128, ROWS / 128), blk, 0, stream>>>(
        xb, kvdwb, kvd_b, latb, ROWS, LATENT, EMBED);
    // kv = latent @ kvu_w^T + kvu_b
    gemm_bias<false><<<dim3((2 * EMBED) / 128, ROWS / 128), blk, 0, stream>>>(
        latb, kvuwb, kvu_b, kvb, ROWS, 2 * EMBED, LATENT);
    // attention
    mla_attn<<<dim3(SEQ / 128, NHEADS, BATCH), blk, 0, stream>>>(qb, kvb, attnb);
    // out = attn @ out_w^T + out_b  (fp32 final output)
    gemm_bias<true><<<dim3(EMBED / 128, ROWS / 128), blk, 0, stream>>>(
        attnb, outwb, out_b, (float*)d_out, ROWS, EMBED, EMBED);
}
